// GCBlockP1_41781441856162
// MI455X (gfx1250) — matrix operand order repaired and checked by reading, not compiled, not run on hardware
//
#include <hip/hip_runtime.h>
#include <hip/hip_bf16.h>

// ---------------------------------------------------------------------------
// CDNA5 / gfx1250 fused GNN block: node MLP -> edge PI layer (basis-contracted)
// -> edge MLP -> scatter-add.  All matmuls via v_wmma_f32_16x16x32_bf16.
// Edge-kernel weight fragments staged in 320KB/WGP LDS; branch-free tanh.
// ---------------------------------------------------------------------------

typedef __attribute__((ext_vector_type(16))) __bf16 v16bf;
typedef __attribute__((ext_vector_type(8)))  __bf16 v8bf;
typedef __attribute__((ext_vector_type(8)))  float  v8f;

#define DD 64
#define NBAS 8
#define NW 8            // waves per block (256 threads, wave32)
#define ST_STRIDE 72    // LDS staging row stride in bf16 elems (144B, 16B aligned)

// fragment-region element offsets (bf16 elems) in global workspace
#define OFF_PPW1 0      // 4x2  frags
#define OFF_PPW2 4096   // 4x2
#define OFF_PIW1 8192   // 4x4
#define OFF_PIW2 16384  // 4x2
#define OFF_IIW1 20480  // 4x2
#define OFF_IIW2 24576  // 4x2
#define OFF_PIW3 28672  // 8(b) x 4(nt) x 2(kt)
#define FRAG_ELEMS 61440

// LDS-local offsets inside edge kernel's fragment cache (same order as global)
#define L_PIW1 0
#define L_PIW2 8192
#define L_IIW1 12288
#define L_IIW2 16384
#define L_PIW3 20480
#define EDGE_FRAG_ELEMS 53248   // 104 KB

// ---------------------------------------------------------------------------

__device__ __forceinline__ v8f wmma_bf16(v16bf a, v16bf b, v8f c) {
  return __builtin_amdgcn_wmma_f32_16x16x32_bf16(false, a, false, b, (short)0, c,
                                                 false, false);
}

__device__ __forceinline__ void wave_lds_fence() {
  __builtin_amdgcn_wave_barrier();
  asm volatile("s_wait_dscnt 0" ::: "memory");
  __builtin_amdgcn_wave_barrier();
}

// branch-free tanh: sign(x) * (1 - 2/(exp(2|x|)+1)); v_exp_f32 + v_rcp_f32
__device__ __forceinline__ float fast_tanh(float x) {
  float ax = __builtin_fabsf(x);
  float e  = __expf(ax + ax);                       // native v_exp path
  float t  = 1.f - 2.f * __builtin_amdgcn_rcpf(e + 1.f);
  return __builtin_copysignf(t, x);
}

__device__ __forceinline__ v16bf cat8(v8bf lo, v8bf hi) {
  v16bf r;
#pragma unroll
  for (int i = 0; i < 8; ++i) { r[i] = lo[i]; r[i + 8] = hi[i]; }
  return r;
}

// A-fragment (16x32 bf16) from a row-major bf16 row of 64; koff = kt*32 + 8*(lane>>4)
__device__ __forceinline__ v16bf afrag_bf16(const __bf16* row, int koff) {
  v8bf lo = *(const v8bf*)(row + koff);
  v8bf hi = *(const v8bf*)(row + koff + 16);
  return cat8(lo, hi);
}

// Same but from an f32 row (node features), converting on the fly
__device__ __forceinline__ v16bf afrag_f32(const float* row, int koff) {
  v8f lo = *(const v8f*)(row + koff);
  v8f hi = *(const v8f*)(row + koff + 16);
  v16bf r;
#pragma unroll
  for (int i = 0; i < 8; ++i) { r[i] = (__bf16)lo[i]; r[i + 8] = (__bf16)hi[i]; }
  return r;
}

// pre-packed B fragment: 32 lanes x 16 bf16, lane-contiguous (works for LDS ptrs)
__device__ __forceinline__ v16bf bfrag(const __bf16* base, int lane) {
  return *(const v16bf*)(base + lane * 16);
}

// store C-layout accumulator (16x16 f32) into LDS staging tile as bf16
__device__ __forceinline__ void stage_store(__bf16* st, int nt, int lane, v8f acc) {
  int col = nt * 16 + (lane & 15);
  int rbase = (lane >> 4) << 3;
#pragma unroll
  for (int g = 0; g < 8; ++g) st[(rbase + g) * ST_STRIDE + col] = (__bf16)acc[g];
}

// read A fragment (kt-th K-step) from LDS staging tile
__device__ __forceinline__ v16bf lds_afrag(const __bf16* st, int kt, int lane) {
  const __bf16* p = st + (lane & 15) * ST_STRIDE + kt * 32 + ((lane >> 4) << 3);
  v8bf lo = *(const v8bf*)p;
  v8bf hi = *(const v8bf*)(p + 16);
  return cat8(lo, hi);
}

// ---------------------------------------------------------------------------
// Weight pre-packing: row-major f32 [K x ncols] -> bf16 B fragments
// B layout: elem e of lane L holds K = kt*32 + (e&7) + 16*(e>=8) + 8*(L>=16), N = nt*16 + (L&15)
// frag index = nt*kts + kt
// ---------------------------------------------------------------------------
__global__ __launch_bounds__(256) void pack_bfrag(const float* __restrict__ W,
                                                  __bf16* __restrict__ dst,
                                                  int kts, int ncols) {
  int tid = blockIdx.x * blockDim.x + threadIdx.x;
  int total = kts * 4 * 512;
  if (tid >= total) return;
  int e = tid & 15;
  int lane = (tid >> 4) & 31;
  int frag = tid >> 9;
  int kt = frag % kts;
  int nt = frag / kts;
  int K = kt * 32 + (e & 7) + ((e >> 3) << 4) + ((lane >> 4) << 3);
  int N = nt * 16 + (lane & 15);
  dst[frag * 512 + lane * 16 + e] = (__bf16)W[K * ncols + N];
}

// W3 [64 x 512] -> 8 per-basis [64x64] matrices W3b[d,c] = W3[d, c*8+b],
// packed as B fragments; frag index = (b*4+nt)*2 + kt
__global__ __launch_bounds__(256) void pack_w3(const float* __restrict__ W3,
                                               __bf16* __restrict__ dst) {
  int tid = blockIdx.x * blockDim.x + threadIdx.x;  // 64*512 = 32768 total
  if (tid >= 32768) return;
  int e = tid & 15;
  int lane = (tid >> 4) & 31;
  int frag = tid >> 9;
  int kt = frag & 1;
  int nt = (frag >> 1) & 3;
  int b = frag >> 3;
  int K = kt * 32 + (e & 7) + ((e >> 3) << 4) + ((lane >> 4) << 3);
  int c = nt * 16 + (lane & 15);
  dst[frag * 512 + lane * 16 + e] = (__bf16)W3[K * 512 + c * 8 + b];
}

__global__ __launch_bounds__(256) void zero_f32(float* __restrict__ o, long n) {
  long i = (long)blockIdx.x * blockDim.x + threadIdx.x;
  if (i < n) o[i] = 0.f;
}

// ---------------------------------------------------------------------------
// Node kernel: p1 = tanh(p@ppW1+b1)@ppW2+b2, stored as bf16 [N x 64]
// one wave per 16-node tile; pp weight frags cached in LDS per block
// ---------------------------------------------------------------------------
__global__ __launch_bounds__(256) void node_kernel(const float* __restrict__ p,
                                                   const __bf16* __restrict__ frags,
                                                   const float* __restrict__ b1,
                                                   const float* __restrict__ b2,
                                                   __bf16* __restrict__ p1bf, int N) {
  __shared__ __attribute__((aligned(16))) __bf16 sfrag[8192];       // 16 KB
  __shared__ __attribute__((aligned(16))) __bf16 st[NW][16 * ST_STRIDE];

  // cooperative fragment load (ppW1 + ppW2, contiguous) then block barrier
#pragma unroll
  for (int i = threadIdx.x; i < 8192 / 8; i += 256)
    ((v8bf*)sfrag)[i] = ((const v8bf*)(frags + OFF_PPW1))[i];
  __syncthreads();

  int lane = threadIdx.x & 31;
  int w = threadIdx.x >> 5;
  int ntiles = (N + 15) >> 4;
  int tile = blockIdx.x * NW + w;
  if (tile >= ntiles) return;
  int n0 = tile << 4;

  int nr = n0 + (lane & 15);
  if (nr >= N) nr = N - 1;
  int ko = (lane >> 4) << 3;

  v16bf a0 = afrag_f32(p + (size_t)nr * DD, ko);
  v16bf a1 = afrag_f32(p + (size_t)nr * DD, 32 + ko);

#pragma unroll
  for (int nt = 0; nt < 4; ++nt) {
    v8f acc = {0.f, 0.f, 0.f, 0.f, 0.f, 0.f, 0.f, 0.f};
    acc = wmma_bf16(a0, bfrag(sfrag + (nt * 2 + 0) * 512, lane), acc);
    acc = wmma_bf16(a1, bfrag(sfrag + (nt * 2 + 1) * 512, lane), acc);
    float bb = b1[nt * 16 + (lane & 15)];
#pragma unroll
    for (int g = 0; g < 8; ++g) acc[g] = fast_tanh(acc[g] + bb);
    stage_store(st[w], nt, lane, acc);
  }
  wave_lds_fence();
  v16bf t0 = lds_afrag(st[w], 0, lane);
  v16bf t1 = lds_afrag(st[w], 1, lane);
  __builtin_amdgcn_wave_barrier();

#pragma unroll
  for (int nt = 0; nt < 4; ++nt) {
    v8f acc = {0.f, 0.f, 0.f, 0.f, 0.f, 0.f, 0.f, 0.f};
    acc = wmma_bf16(t0, bfrag(sfrag + 4096 + (nt * 2 + 0) * 512, lane), acc);
    acc = wmma_bf16(t1, bfrag(sfrag + 4096 + (nt * 2 + 1) * 512, lane), acc);
    float bb = b2[nt * 16 + (lane & 15)];
    int rbase = (lane >> 4) << 3;
#pragma unroll
    for (int g = 0; g < 8; ++g) {
      int row = n0 + rbase + g;
      if (row < N)
        p1bf[(size_t)row * DD + nt * 16 + (lane & 15)] = (__bf16)(acc[g] + bb);
    }
  }
}

// ---------------------------------------------------------------------------
// Edge kernel: one wave per 16-edge tile, fully fused PI + basis + ii + scatter
// ALL weight fragments (104 KB) cached in LDS per block (320 KB/WGP budget)
// ---------------------------------------------------------------------------
__global__ __launch_bounds__(256) void edge_kernel(
    const __bf16* __restrict__ p1bf, const int* __restrict__ idx_i,
    const int* __restrict__ idx_j, const float* __restrict__ basis,
    const __bf16* __restrict__ frags, const float* __restrict__ pib1,
    const float* __restrict__ pib2, const float* __restrict__ iib1,
    const float* __restrict__ iib2, float* __restrict__ out, int E) {
  __shared__ __attribute__((aligned(16))) __bf16 sfrag[EDGE_FRAG_ELEMS];  // 104 KB
  __shared__ __attribute__((aligned(16))) __bf16 st[NW][16 * ST_STRIDE];  // 18 KB
  __shared__ __attribute__((aligned(16))) float bas[NW][128];             // 4 KB

  // cooperative fragment load (piW1,piW2,iiW1,iiW2,piW3 contiguous in ws)
#pragma unroll 2
  for (int i = threadIdx.x; i < EDGE_FRAG_ELEMS / 8; i += 256)
    ((v8bf*)sfrag)[i] = ((const v8bf*)(frags + OFF_PIW1))[i];
  __syncthreads();

  int lane = threadIdx.x & 31;
  int w = threadIdx.x >> 5;
  int ntiles = (E + 15) >> 4;
  int tile = blockIdx.x * NW + w;
  if (tile >= ntiles) return;
  int e0 = tile << 4;

  int er = e0 + (lane & 15);
  if (er >= E) er = E - 1;
  int gi = idx_i[er];
  int gj = idx_j[er];

  int rbase = (lane >> 4) << 3;
  int jrow[8];
#pragma unroll
  for (int g = 0; g < 8; ++g) {
    int e = e0 + rbase + g;
    if (e >= E) e = E - 1;
    jrow[g] = idx_j[e];
  }

  // basis tile (16 rows x 8) -> LDS, coalesced float4
  ((float4*)bas[w])[lane] = ((const float4*)(basis + (size_t)e0 * NBAS))[lane];

  // A fragments for concat(p1[i], p1[j])  (K = 0..127, 4 K-steps)
  int ko = (lane >> 4) << 3;
  v16bf ah[4];
  ah[0] = afrag_bf16(p1bf + (size_t)gi * DD, ko);
  ah[1] = afrag_bf16(p1bf + (size_t)gi * DD, 32 + ko);
  ah[2] = afrag_bf16(p1bf + (size_t)gj * DD, ko);
  ah[3] = afrag_bf16(p1bf + (size_t)gj * DD, 32 + ko);

  wave_lds_fence();  // basis tile visible to all lanes of this wave

  // ---- t1 = h @ piW1 + b1 -------------------------------------------------
#pragma unroll
  for (int nt = 0; nt < 4; ++nt) {
    v8f acc = {0.f, 0.f, 0.f, 0.f, 0.f, 0.f, 0.f, 0.f};
#pragma unroll
    for (int kt = 0; kt < 4; ++kt)
      acc = wmma_bf16(ah[kt], bfrag(sfrag + L_PIW1 + (nt * 4 + kt) * 512, lane), acc);
    float bb = pib1[nt * 16 + (lane & 15)];
#pragma unroll
    for (int g = 0; g < 8; ++g) acc[g] += bb;
    stage_store(st[w], nt, lane, acc);
  }
  wave_lds_fence();
  v16bf a1k0 = lds_afrag(st[w], 0, lane);
  v16bf a1k1 = lds_afrag(st[w], 1, lane);
  __builtin_amdgcn_wave_barrier();

  // ---- t2 = t1 @ piW2 + b2 ------------------------------------------------
#pragma unroll
  for (int nt = 0; nt < 4; ++nt) {
    v8f acc = {0.f, 0.f, 0.f, 0.f, 0.f, 0.f, 0.f, 0.f};
    acc = wmma_bf16(a1k0, bfrag(sfrag + L_PIW2 + (nt * 2 + 0) * 512, lane), acc);
    acc = wmma_bf16(a1k1, bfrag(sfrag + L_PIW2 + (nt * 2 + 1) * 512, lane), acc);
    float bb = pib2[nt * 16 + (lane & 15)];
#pragma unroll
    for (int g = 0; g < 8; ++g) acc[g] += bb;
    stage_store(st[w], nt, lane, acc);
  }
  wave_lds_fence();
  v16bf a2k0 = lds_afrag(st[w], 0, lane);
  v16bf a2k1 = lds_afrag(st[w], 1, lane);
  __builtin_amdgcn_wave_barrier();

  // ---- i1 = einsum over basis of t2 @ W3  (64 WMMA, basis as row scaling) --
  v8f i1a[4];
#pragma unroll
  for (int nt = 0; nt < 4; ++nt)
#pragma unroll
    for (int g = 0; g < 8; ++g) i1a[nt][g] = 0.f;

#pragma unroll
  for (int b = 0; b < NBAS; ++b) {
    float sb[8];
#pragma unroll
    for (int g = 0; g < 8; ++g) sb[g] = bas[w][(rbase + g) * NBAS + b];
#pragma unroll
    for (int nt = 0; nt < 4; ++nt) {
      v8f acc = {0.f, 0.f, 0.f, 0.f, 0.f, 0.f, 0.f, 0.f};
      acc = wmma_bf16(a2k0, bfrag(sfrag + L_PIW3 + ((b * 4 + nt) * 2 + 0) * 512, lane), acc);
      acc = wmma_bf16(a2k1, bfrag(sfrag + L_PIW3 + ((b * 4 + nt) * 2 + 1) * 512, lane), acc);
#pragma unroll
      for (int g = 0; g < 8; ++g) i1a[nt][g] = fmaf(sb[g], acc[g], i1a[nt][g]);
    }
  }

  // ---- ii layer: tanh(i1@iiW1+b1)@iiW2+b2 ---------------------------------
#pragma unroll
  for (int nt = 0; nt < 4; ++nt) stage_store(st[w], nt, lane, i1a[nt]);
  wave_lds_fence();
  v16bf a3k0 = lds_afrag(st[w], 0, lane);
  v16bf a3k1 = lds_afrag(st[w], 1, lane);
  __builtin_amdgcn_wave_barrier();

#pragma unroll
  for (int nt = 0; nt < 4; ++nt) {
    v8f acc = {0.f, 0.f, 0.f, 0.f, 0.f, 0.f, 0.f, 0.f};
    acc = wmma_bf16(a3k0, bfrag(sfrag + L_IIW1 + (nt * 2 + 0) * 512, lane), acc);
    acc = wmma_bf16(a3k1, bfrag(sfrag + L_IIW1 + (nt * 2 + 1) * 512, lane), acc);
    float bb = iib1[nt * 16 + (lane & 15)];
#pragma unroll
    for (int g = 0; g < 8; ++g) acc[g] = fast_tanh(acc[g] + bb);
    stage_store(st[w], nt, lane, acc);
  }
  wave_lds_fence();
  v16bf a4k0 = lds_afrag(st[w], 0, lane);
  v16bf a4k1 = lds_afrag(st[w], 1, lane);
  __builtin_amdgcn_wave_barrier();

#pragma unroll
  for (int nt = 0; nt < 4; ++nt) {
    v8f acc = {0.f, 0.f, 0.f, 0.f, 0.f, 0.f, 0.f, 0.f};
    acc = wmma_bf16(a4k0, bfrag(sfrag + L_IIW2 + (nt * 2 + 0) * 512, lane), acc);
    acc = wmma_bf16(a4k1, bfrag(sfrag + L_IIW2 + (nt * 2 + 1) * 512, lane), acc);
    float bb = iib2[nt * 16 + (lane & 15)];
    int col = nt * 16 + (lane & 15);
#pragma unroll
    for (int g = 0; g < 8; ++g) {
      int e = e0 + rbase + g;
      if (e < E) atomicAdd(out + (size_t)jrow[g] * DD + col, acc[g] + bb);
    }
  }
}

// ---------------------------------------------------------------------------
extern "C" void kernel_launch(void* const* d_in, const int* in_sizes, int n_in,
                              void* d_out, int out_size, void* d_ws, size_t ws_size,
                              hipStream_t stream) {
  const float* p     = (const float*)d_in[0];
  const int*   idx_i = (const int*)d_in[1];
  const int*   idx_j = (const int*)d_in[2];
  const float* basis = (const float*)d_in[3];
  const float* ppW1  = (const float*)d_in[4];
  const float* ppb1  = (const float*)d_in[5];
  const float* ppW2  = (const float*)d_in[6];
  const float* ppb2  = (const float*)d_in[7];
  const float* piW1  = (const float*)d_in[8];
  const float* pib1  = (const float*)d_in[9];
  const float* piW2  = (const float*)d_in[10];
  const float* pib2  = (const float*)d_in[11];
  const float* piW3  = (const float*)d_in[12];
  const float* iiW1  = (const float*)d_in[13];
  const float* iib1  = (const float*)d_in[14];
  const float* iiW2  = (const float*)d_in[15];
  const float* iib2  = (const float*)d_in[16];

  int N = in_sizes[0] / DD;  // 50000
  int E = in_sizes[1];       // 800000
  float* out = (float*)d_out;

  __bf16* p1bf  = (__bf16*)d_ws;
  __bf16* frags = p1bf + (size_t)N * DD;

  // 1) pack weights into WMMA B-fragments (bf16)
  pack_bfrag<<<16, 256, 0, stream>>>(ppW1, frags + OFF_PPW1, 2, DD);
  pack_bfrag<<<16, 256, 0, stream>>>(ppW2, frags + OFF_PPW2, 2, DD);
  pack_bfrag<<<32, 256, 0, stream>>>(piW1, frags + OFF_PIW1, 4, DD);
  pack_bfrag<<<16, 256, 0, stream>>>(piW2, frags + OFF_PIW2, 2, DD);
  pack_bfrag<<<16, 256, 0, stream>>>(iiW1, frags + OFF_IIW1, 2, DD);
  pack_bfrag<<<16, 256, 0, stream>>>(iiW2, frags + OFF_IIW2, 2, DD);
  pack_w3<<<128, 256, 0, stream>>>(piW3, frags + OFF_PIW3);

  // 2) zero the scatter target
  long outn = (long)N * DD;
  zero_f32<<<(int)((outn + 255) / 256), 256, 0, stream>>>(out, outn);

  // 3) node MLP -> p1 (bf16)
  int ntiles_n = (N + 15) >> 4;
  node_kernel<<<(ntiles_n + NW - 1) / NW, 256, 0, stream>>>(p, frags, ppb1, ppb2,
                                                            p1bf, N);

  // 4) fused edge pipeline + scatter
  int ntiles_e = (E + 15) >> 4;
  edge_kernel<<<(ntiles_e + NW - 1) / NW, 256, 0, stream>>>(
      p1bf, idx_i, idx_j, basis, frags, pib1, pib2, iib1, iib2, out, E);
}
